// DiagonalElmanCell_29978871726682
// MI455X (gfx1250) — matrix operand order repaired and checked by reading, not compiled, not run on hardware
//
#include <hip/hip_runtime.h>

// ---- problem dims (fixed by the reference) ----
#define T_DIM 1024
#define B_DIM 16
#define D_DIM 1024
#define M_DIM (T_DIM * B_DIM)   // 16384 rows of the GEMM

typedef __attribute__((ext_vector_type(16))) __bf16 v16bf;
typedef __attribute__((ext_vector_type(8)))  __bf16 v8bf;   // 16 bytes -> b128
typedef __attribute__((ext_vector_type(4)))  __bf16 v4bf;   // 8 bytes
typedef __attribute__((ext_vector_type(8)))  float  v8f;

struct bf16x16 { v8bf lo, hi; };

// fp32 -> bf16, round-to-nearest-even
static __device__ __forceinline__ __bf16 f2bf(float f) {
    unsigned u = __builtin_bit_cast(unsigned, f);
    unsigned r = u + 0x7FFFu + ((u >> 16) & 1u);
    unsigned short h = (unsigned short)(r >> 16);
    return __builtin_bit_cast(__bf16, h);
}

// tanh(z) = 1 - 2/(1 + e^{2z});  correct limits: z->+inf -> 1, z->-inf -> -1
static __device__ __forceinline__ float tanh_fast(float z) {
    return 1.0f - 2.0f / (1.0f + __expf(2.0f * z));
}

// -----------------------------------------------------------------------------
// Pre-pass: one-time fp32 -> bf16 conversion (float4 in, 4x bf16 out).
// -----------------------------------------------------------------------------
__global__ __launch_bounds__(256)
void cvt_bf16_kernel(const float* __restrict__ src, __bf16* __restrict__ dst,
                     int n4) {
    const int i = blockIdx.x * blockDim.x + threadIdx.x;
    if (i >= n4) return;
    const float4 v = ((const float4*)src)[i];
    v4bf o;
    o[0] = f2bf(v.x); o[1] = f2bf(v.y); o[2] = f2bf(v.z); o[3] = f2bf(v.w);
    ((v4bf*)dst)[i] = o;
}

// -----------------------------------------------------------------------------
// GEMM: x_proj = X @ W^T + b via v_wmma_f32_16x16x32_bf16.
// One wave -> 32x64 tile: 2 A-frags, 4 shared B-frags, 8 accumulators per
// K-step => 12 b128 loads per 8 WMMAs and ~1.6 GB total L2 operand traffic.
// Row m result lands in hbuf[(m + B_DIM)*D_DIM + n] (the h[t+1] slot).
// -----------------------------------------------------------------------------
__global__ __launch_bounds__(256)
void xproj_wmma_kernel(const __bf16* __restrict__ xbf,
                       const __bf16* __restrict__ wbf,
                       const float* __restrict__ bias,
                       float* __restrict__ hbuf) {
    const int lane  = threadIdx.x & 31;
    const int wave  = blockIdx.x * (blockDim.x >> 5) + (threadIdx.x >> 5);
    const int strips    = D_DIM / 64;         // 16 column strips
    const int row_tile  = wave / strips;      // 0..511 (32 rows each)
    const int col_strip = wave % strips;      // 0..15  (64 cols each)
    const int half = lane >> 4;
    const int l15  = lane & 15;

    // A (16x32 bf16): lane holds row l15; K = half*8+[0,8) and 16+half*8+[0,8).
    const __bf16* xptr0 = xbf + (size_t)(row_tile * 32 + l15) * D_DIM + half * 8;
    const __bf16* xptr1 = xptr0 + (size_t)16 * D_DIM;

    // B (32x16 bf16): lane holds column n => row n of W; K = half*8+[0,8) and 16+half*8+[0,8).
    const __bf16* wptr[4];
#pragma unroll
    for (int s = 0; s < 4; ++s)
        wptr[s] = wbf + (size_t)(col_strip * 64 + s * 16 + l15) * D_DIM
                      + half * 8;

    v8f acc0[4], acc1[4];
#pragma unroll
    for (int s = 0; s < 4; ++s) {
        acc0[s] = (v8f){0.f,0.f,0.f,0.f,0.f,0.f,0.f,0.f};
        acc1[s] = (v8f){0.f,0.f,0.f,0.f,0.f,0.f,0.f,0.f};
    }

    for (int k0 = 0; k0 < D_DIM; k0 += 32) {
        const v8bf a0l = *(const v8bf*)(xptr0 + k0);
        const v8bf a0h = *(const v8bf*)(xptr0 + k0 + 16);
        const v16bf a0 = __builtin_bit_cast(v16bf, (bf16x16){a0l, a0h});
        const v8bf a1l = *(const v8bf*)(xptr1 + k0);
        const v8bf a1h = *(const v8bf*)(xptr1 + k0 + 16);
        const v16bf a1 = __builtin_bit_cast(v16bf, (bf16x16){a1l, a1h});
#pragma unroll
        for (int s = 0; s < 4; ++s) {
            const v8bf b0 = *(const v8bf*)(wptr[s] + k0);
            const v8bf b1 = *(const v8bf*)(wptr[s] + k0 + 16);
            const v16bf bfrag = __builtin_bit_cast(v16bf, (bf16x16){b0, b1});
            acc0[s] = __builtin_amdgcn_wmma_f32_16x16x32_bf16(
                false, a0, false, bfrag, (short)0, acc0[s], false, false);
            acc1[s] = __builtin_amdgcn_wmma_f32_16x16x32_bf16(
                false, a1, false, bfrag, (short)0, acc1[s], false, false);
        }
    }

    // C/D layout: lane l holds column (l&15), rows 8*(l>>4)+r in acc[.][r].
#pragma unroll
    for (int s = 0; s < 4; ++s) {
        const int n    = col_strip * 64 + s * 16 + l15;
        const float bn = bias[n];
        const int row0 = row_tile * 32 + half * 8;
#pragma unroll
        for (int r = 0; r < 8; ++r) {
            hbuf[(size_t)(row0 + r      + B_DIM) * D_DIM + n] = acc0[s][r] + bn;
            hbuf[(size_t)(row0 + r + 16 + B_DIM) * D_DIM + n] = acc1[s][r] + bn;
        }
    }
}

// -----------------------------------------------------------------------------
// Diagonal scan + SiLU gate, software-pipelined: next timestep's x/xp loads
// are issued before the dependent tanh/store so memory latency overlaps the
// serial recurrence. One thread per (b,d) chain; coalesced at every t.
// -----------------------------------------------------------------------------
__global__ __launch_bounds__(256)
void scan_gate_kernel(const float* __restrict__ x,
                      const float* __restrict__ h0,
                      const float* __restrict__ alpha_raw,
                      const float* __restrict__ b_gate,
                      float* __restrict__ out,
                      float* __restrict__ hbuf) {
    const int idx = blockIdx.x * blockDim.x + threadIdx.x;   // 0 .. B*D-1
    if (idx >= B_DIM * D_DIM) return;
    const int d = idx & (D_DIM - 1);

    const float alpha = 1.0f / (1.0f + __expf(-alpha_raw[d]));
    const float bg    = b_gate[d];
    const int stride  = B_DIM * D_DIM;

    float h = h0[idx];
    hbuf[idx] = h;                                           // h[0] = h0

    float xv = x[idx];                                       // t = 0 operands
    float xp = hbuf[(size_t)stride + idx];

    for (int t = 0; t < T_DIM; ++t) {
        const size_t o = (size_t)t * stride + idx;
        // issue next iteration's loads before the dependent math
        float xv_n = 0.f, xp_n = 0.f;
        if (t + 1 < T_DIM) {
            xv_n = x[o + stride];
            xp_n = hbuf[o + 2 * (size_t)stride];
        }
        h = tanh_fast(xp + alpha * h);
        hbuf[(size_t)stride + o] = h;                        // h[t+1]
        const float z = xv + bg;
        out[o] = h * (z / (1.0f + __expf(-z)));              // h * silu(x+bg)
        xv = xv_n;
        xp = xp_n;
    }
}

extern "C" void kernel_launch(void* const* d_in, const int* in_sizes, int n_in,
                              void* d_out, int out_size, void* d_ws, size_t ws_size,
                              hipStream_t stream) {
    const float* x         = (const float*)d_in[0];   // [T,B,D]
    const float* h0        = (const float*)d_in[1];   // [B,D]
    const float* W_x       = (const float*)d_in[2];   // [D,D]
    const float* alpha_raw = (const float*)d_in[3];   // [D]
    const float* bias      = (const float*)d_in[4];   // [D]
    const float* b_gate    = (const float*)d_in[5];   // [D]

    float* out  = (float*)d_out;                          // [T,B,D]
    float* hbuf = out + (size_t)T_DIM * B_DIM * D_DIM;    // [T+1,B,D]

    // Stage bf16 copies of W and X in the (currently unused) `out` region:
    // Wbf: 2 MB, Xbf: 32 MB -- both overwritten later by the scan kernel.
    __bf16* wbf = (__bf16*)out;
    __bf16* xbf = wbf + (size_t)D_DIM * D_DIM;

    const int w4 = (D_DIM * D_DIM) / 4;
    cvt_bf16_kernel<<<(w4 + 255) / 256, 256, 0, stream>>>(W_x, wbf, w4);
    const int x4 = (M_DIM * D_DIM) / 4;
    cvt_bf16_kernel<<<(x4 + 255) / 256, 256, 0, stream>>>(x, xbf, x4);

    // GEMM: 512 row-tiles x 16 col-strips = 8192 waves, 8 waves/block.
    const int total_waves = (M_DIM / 32) * (D_DIM / 64);
    xproj_wmma_kernel<<<total_waves / 8, 256, 0, stream>>>(xbf, wbf, bias, hbuf);

    // Scan: B*D = 16384 independent chains.
    const int n = B_DIM * D_DIM;
    scan_gate_kernel<<<(n + 255) / 256, 256, 0, stream>>>(x, h0, alpha_raw,
                                                          b_gate, out, hbuf);
}
